// MultiHeadAttentionEncoder_78219944395391
// MI455X (gfx1250) — matrix operand order repaired and checked by reading, not compiled, not run on hardware
//
#include <hip/hip_runtime.h>
#include <hip/hip_bf16.h>
#include <math.h>

#define DIMX 1024
#define NHEAD 16
#define HD 64
#define SEQ 2048
#define NBATCH 4
#define TK 32   // keys per attention iteration

typedef __bf16 v16bf __attribute__((ext_vector_type(16)));
typedef __bf16 v8bf  __attribute__((ext_vector_type(8)));
typedef float  v8f   __attribute__((ext_vector_type(8)));
typedef unsigned int u32x4 __attribute__((ext_vector_type(4)));
typedef int i32x4 __attribute__((ext_vector_type(4)));
typedef int i32x8 __attribute__((ext_vector_type(8)));

#if defined(__has_builtin)
#if __has_builtin(__builtin_amdgcn_tensor_load_to_lds) && \
    __has_builtin(__builtin_amdgcn_s_wait_tensorcnt)
#define USE_TDM 1
#endif
#endif
#ifndef USE_TDM
#define USE_TDM 0
#endif

static __device__ __forceinline__ v8f wmma_bf16(v16bf a, v16bf b, v8f c) {
    // D = A(16x32 bf16) * B(32x16 bf16) + C(16x16 f32)
    return __builtin_amdgcn_wmma_f32_16x16x32_bf16(
        /*neg_a=*/false, a, /*neg_b=*/false, b,
        /*c_mod=*/(short)0, c, /*reuse_a=*/false, /*reuse_b=*/false);
}

// A-fragment (16x32 bf16) from two contiguous 8-element runs:
// K = base + e for e = 0..15  (base = half*16, p1 = p0 + 8)
static __device__ __forceinline__ v16bf frag16(const __bf16* p0, const __bf16* p1) {
    v8bf lo = *(const v8bf*)p0;
    v8bf hi = *(const v8bf*)p1;
    return __builtin_shufflevector(lo, hi, 0, 1, 2, 3, 4, 5, 6, 7,
                                           8, 9, 10, 11, 12, 13, 14, 15);
}

#if USE_TDM
// Issue one TDM descriptor: load a [32 x 64] bf16 tile (row stride 64) from a
// [SEQ x 64] bf16 tensor at gsrc into LDS byte offset lds_off (contiguous).
// D# layout per CDNA5 ISA.
// This toolchain exposes the 6-arg builtin:
//   (u32x4 g0, i32x8 g1, i32x4 g2, i32x4 g3, i32x8 extra, i32 cpol)
static __device__ __forceinline__ void tdm_load_tile_32x64(const __bf16* gsrc,
                                                           unsigned lds_off) {
    const unsigned long long ga = (unsigned long long)(uintptr_t)gsrc;
    u32x4 g0;
    g0[0] = 1u;                                   // count=1 (valid), user mode
    g0[1] = lds_off;                              // lds_addr (bytes)
    g0[2] = (unsigned)(ga & 0xffffffffu);         // global_addr[31:0]
    g0[3] = (unsigned)((ga >> 32) & 0x01ffffffu)  // global_addr[56:32]
          | 0x80000000u;                          // type = 2 ("image")
    i32x8 g1;
    g1[0] = (int)(1u << 16);        // workgroup_mask=0 | data_size=1 (2 bytes)
    g1[1] = (int)(64u << 16);       // tensor_dim0[15:0] = 64  (bits 63:48)
    g1[2] = (int)(2048u << 16);     // tensor_dim1[15:0] = SEQ (bits 95:80)
    g1[3] = (int)(64u << 16);       // tile_dim0 = 64          (bits 127:112)
    g1[4] = 32;                     // tile_dim1 = 32
    g1[5] = 64;                     // tensor_dim0_stride = 64 elements
    g1[6] = 0;
    g1[7] = 0;
    const i32x4 z4 = {0, 0, 0, 0};            // groups 2/3 unused (2D tensor)
    const i32x8 z8 = {0, 0, 0, 0, 0, 0, 0, 0};
    __builtin_amdgcn_tensor_load_to_lds(g0, g1, z4, z4, z8, 0);
}
#endif

// ---------------------------------------------------------------------------
// Kernel 1: per-head QKV projection  y = x @ W^T + b, stored bf16 [H, S, 64]
// grid = (SEQ/16, NHEAD), block = 128 (4 waves); wave w owns output cols
// [16w, 16w+16) and computes q, k, v tiles (2 WMMAs each, K=64).
// ---------------------------------------------------------------------------
__global__ void __launch_bounds__(128)
qkv_project_kernel(const float* __restrict__ x,     // [SEQ, DIMX] (last batch)
                   const float* __restrict__ Wq, const float* __restrict__ bq,
                   const float* __restrict__ Wk, const float* __restrict__ bk,
                   const float* __restrict__ Wv, const float* __restrict__ bv,
                   __bf16* __restrict__ qo, __bf16* __restrict__ ko,
                   __bf16* __restrict__ vo)
{
    const int h    = blockIdx.y;
    const int row0 = blockIdx.x * 16;
    const int lane = threadIdx.x & 31;
    const int wave = threadIdx.x >> 5;       // col tile 0..3
    const int n    = lane & 15;
    const int half = lane >> 4;              // 0 / 1

    // --- A fragments from x: row = row0 + m (m = lane%16), col = h*64 + k ---
    v16bf a[2];
    const float* xrow = x + (size_t)(row0 + n) * DIMX + h * HD;
#pragma unroll
    for (int ks = 0; ks < 2; ++ks) {
        const float* p0 = xrow + ks * 32 + half * 16;
        const float* p1 = p0 + 8;
#pragma unroll
        for (int e = 0; e < 8; ++e) {
            a[ks][e]     = (__bf16)p0[e];
            a[ks][e + 8] = (__bf16)p1[e];
        }
    }

    const float* Ws[3]   = {Wq, Wk, Wv};
    const float* bs[3]   = {bq, bk, bv};
    __bf16*      outs[3] = {qo, ko, vo};
    const int ng = wave * 16 + n;            // output feature within head

#pragma unroll
    for (int mtx = 0; mtx < 3; ++mtx) {
        // B[k][n] = W[h][ng][k]  -> contiguous along k per lane
        const float* W = Ws[mtx] + ((size_t)h * HD + ng) * HD;
        const float bval = bs[mtx][h * HD + ng];
        v8f acc;
#pragma unroll
        for (int j = 0; j < 8; ++j) acc[j] = bval;   // C(m,n) = b[n]

#pragma unroll
        for (int ks = 0; ks < 2; ++ks) {
            v16bf b;
            const float* wp = W + ks * 32 + half * 16;  // K = ks*32+half*16+e
#pragma unroll
            for (int e = 0; e < 16; ++e) b[e] = (__bf16)wp[e];
            acc = wmma_bf16(a[ks], b, acc);
        }

        // store D: row = row0 + j + half*8, col = ng
        __bf16* o = outs[mtx] + ((size_t)h * SEQ + row0) * HD + ng;
#pragma unroll
        for (int j = 0; j < 8; ++j)
            o[(size_t)(j + half * 8) * HD] = (__bf16)acc[j];
    }
}

// ---------------------------------------------------------------------------
// Kernel 2: flash attention per head.
// grid = (SEQ/64, NHEAD), block = 128 (4 waves); wave w owns query rows
// [64*bx + 16w, +16). K/V chunks staged in LDS by the TDM (fallback:
// cooperative b128 loads); online softmax; P D->A transpose via LDS.
// ---------------------------------------------------------------------------
#define K_OFF 0                        // bytes: K tile  [32][64] bf16 = 4096
#define V_OFF (TK * HD * 2)            // bytes: V tile  [32][64] bf16 = 4096
#define P_OFF (2 * TK * HD * 2)        // bytes: P tiles [4][16][32] bf16

__global__ void __launch_bounds__(128)
flash_attn_kernel(const __bf16* __restrict__ q, const __bf16* __restrict__ k,
                  const __bf16* __restrict__ v, float* __restrict__ out)
{
    // single arena so static byte offsets match the TDM descriptor lds_addr
    __shared__ __attribute__((aligned(32))) __bf16 smem[2 * TK * HD + 4 * 16 * TK];
    __bf16* Ksm = smem;                        // [TK][HD]
    __bf16* Vsm = smem + TK * HD;              // [TK][HD]

    const int h    = blockIdx.y;
    const int lane = threadIdx.x & 31;
    const int wave = threadIdx.x >> 5;
    const int n    = lane & 15;
    const int half = lane >> 4;
    const int q0   = blockIdx.x * 64 + wave * 16;

    const __bf16* qh = q + (size_t)h * SEQ * HD;
    const __bf16* kh = k + (size_t)h * SEQ * HD;
    const __bf16* vh = v + (size_t)h * SEQ * HD;

    // --- Q A-fragments, resident for the entire key loop ---
    v16bf aq[2];
    {
        const __bf16* qrow = qh + (size_t)(q0 + n) * HD;
        aq[0] = frag16(qrow + half * 16,      qrow + half * 16 + 8);
        aq[1] = frag16(qrow + 32 + half * 16, qrow + 32 + half * 16 + 8);
    }

    v8f acc[4];                       // 16x64 output tile (4 N-subtiles)
#pragma unroll
    for (int es = 0; es < 4; ++es)
#pragma unroll
        for (int j = 0; j < 8; ++j) acc[es][j] = 0.0f;
    float mrow[8], lrow[8];
#pragma unroll
    for (int j = 0; j < 8; ++j) { mrow[j] = -1e30f; lrow[j] = 0.0f; }

    const int ldr = threadIdx.x >> 2;          // 0..31 key row (fallback path)
    const int ldc = (threadIdx.x & 3) * 16;    // 0/16/32/48

    for (int t0 = 0; t0 < SEQ; t0 += TK) {
        __syncthreads();   // previous iteration's LDS reads complete
#if USE_TDM
        if (wave == 0)      tdm_load_tile_32x64(kh + (size_t)t0 * HD, K_OFF);
        else if (wave == 1) tdm_load_tile_32x64(vh + (size_t)t0 * HD, V_OFF);
        else if (t0 + TK < SEQ) {   // idle waves pull next chunk toward L2
            __builtin_prefetch(kh + (size_t)(t0 + TK + ldr) * HD + ldc, 0, 3);
            __builtin_prefetch(vh + (size_t)(t0 + TK + ldr) * HD + ldc, 0, 3);
        }
        if (wave < 2) __builtin_amdgcn_s_wait_tensorcnt(0);
#else
        {
            const uint4* sk = (const uint4*)(kh + (size_t)(t0 + ldr) * HD + ldc);
            const uint4* sv = (const uint4*)(vh + (size_t)(t0 + ldr) * HD + ldc);
            *(uint4*)(Ksm + ldr * HD + ldc)     = sk[0];
            *(uint4*)(Ksm + ldr * HD + ldc + 8) = sk[1];
            *(uint4*)(Vsm + ldr * HD + ldc)     = sv[0];
            *(uint4*)(Vsm + ldr * HD + ldc + 8) = sv[1];
            if (t0 + TK < SEQ) {
                __builtin_prefetch(kh + (size_t)(t0 + TK + ldr) * HD + ldc, 0, 3);
                __builtin_prefetch(vh + (size_t)(t0 + TK + ldr) * HD + ldc, 0, 3);
            }
        }
#endif
        __syncthreads();

        // --- S = Q @ K^T * scale : two 16-key subtiles ---
        v8f sc[2];
#pragma unroll
        for (int nsub = 0; nsub < 2; ++nsub) {
            v8f s;
#pragma unroll
            for (int j = 0; j < 8; ++j) s[j] = 0.0f;
#pragma unroll
            for (int ks = 0; ks < 2; ++ks) {
                // B[k=headdim][n=key]: 16 contiguous bf16 of a K row in LDS
                const __bf16* kp = Ksm + (nsub * 16 + n) * HD + ks * 32 + half * 16;
                s = wmma_bf16(aq[ks], *(const v16bf*)kp, s);
            }
#pragma unroll
            for (int j = 0; j < 8; ++j) s[j] *= 0.125f;   // 1/sqrt(64)
            sc[nsub] = s;
        }

        // --- online softmax (rows live across 16 lanes of half-wave) ---
#pragma unroll
        for (int j = 0; j < 8; ++j) {
            float cm = fmaxf(sc[0][j], sc[1][j]);
            cm = fmaxf(cm, __shfl_xor(cm, 1, 16));
            cm = fmaxf(cm, __shfl_xor(cm, 2, 16));
            cm = fmaxf(cm, __shfl_xor(cm, 4, 16));
            cm = fmaxf(cm, __shfl_xor(cm, 8, 16));
            const float mn    = fmaxf(mrow[j], cm);
            const float alpha = __expf(mrow[j] - mn);
            const float p0 = __expf(sc[0][j] - mn);
            const float p1 = __expf(sc[1][j] - mn);
            float rs = p0 + p1;
            rs += __shfl_xor(rs, 1, 16);
            rs += __shfl_xor(rs, 2, 16);
            rs += __shfl_xor(rs, 4, 16);
            rs += __shfl_xor(rs, 8, 16);
            lrow[j] = lrow[j] * alpha + rs;
            mrow[j] = mn;
            sc[0][j] = p0; sc[1][j] = p1;
#pragma unroll
            for (int es = 0; es < 4; ++es) acc[es][j] *= alpha;
        }

        // --- D-layout P -> A-layout via per-wave LDS tile ---
        __bf16* Pw = smem + (P_OFF / 2) + wave * (16 * TK);
#pragma unroll
        for (int j = 0; j < 8; ++j) {
            Pw[(j + half * 8) * TK + n]      = (__bf16)sc[0][j];
            Pw[(j + half * 8) * TK + 16 + n] = (__bf16)sc[1][j];
        }
        __syncthreads();

        const v16bf ap = frag16(Pw + n * TK + half * 16,
                                Pw + n * TK + half * 16 + 8);

        // --- O += P @ V : 4 N-subtiles over head dim ---
#pragma unroll
        for (int es = 0; es < 4; ++es) {
            v16bf bv;   // B[k=key][n=headdim]: V[half*16 + e][es*16 + n]
#pragma unroll
            for (int e = 0; e < 16; ++e)
                bv[e] = Vsm[(half * 16 + e) * HD + es * 16 + n];
            acc[es] = wmma_bf16(ap, bv, acc[es]);
        }
    }

    // --- normalize and store: out[s][h*64 + e] (hstacked heads) ---
#pragma unroll
    for (int j = 0; j < 8; ++j) {
        const float invl = 1.0f / lrow[j];
        float* orow = out + (size_t)(q0 + j + half * 8) * DIMX + h * HD;
#pragma unroll
        for (int es = 0; es < 4; ++es)
            orow[es * 16 + n] = acc[es][j] * invl;
    }
}

// ---------------------------------------------------------------------------
extern "C" void kernel_launch(void* const* d_in, const int* in_sizes, int n_in,
                              void* d_out, int out_size, void* d_ws, size_t ws_size,
                              hipStream_t stream) {
    (void)in_sizes; (void)n_in; (void)out_size; (void)ws_size;
    const float* x  = (const float*)d_in[0] + (size_t)(NBATCH - 1) * SEQ * DIMX;
    const float* Wq = (const float*)d_in[1];
    const float* bq = (const float*)d_in[2];
    const float* Wk = (const float*)d_in[3];
    const float* bk = (const float*)d_in[4];
    const float* Wv = (const float*)d_in[5];
    const float* bv = (const float*)d_in[6];
    float* out = (float*)d_out;

    const size_t per = (size_t)NHEAD * SEQ * HD;   // 2M elems = 4 MB bf16
    __bf16* qw = (__bf16*)d_ws;                    // needs 12 MB workspace total
    __bf16* kw = qw + per;
    __bf16* vw = kw + per;

    dim3 g1(SEQ / 16, NHEAD), b1(128);
    qkv_project_kernel<<<g1, b1, 0, stream>>>(x, Wq, bq, Wk, bk, Wv, bv, qw, kw, vw);

    dim3 g2(SEQ / 64, NHEAD), b2(128);
    flash_attn_kernel<<<g2, b2, 0, stream>>>(qw, kw, vw, out);
}
